// BasicGRUBlock_21526376087998
// MI455X (gfx1250) — compile-verified, matrix operand order checked by reading, not run on hardware
//
#include <hip/hip_runtime.h>
#include <hip/hip_bf16.h>
#include <stdint.h>

// ---------------------------------------------------------------------------
// BasicGRUBlock for MI455X (gfx1250, wave32, WMMA).
//
//   1. pack weights -> bf16, WMMA-fragment-native layout
//   2. xb = bf16(y + z)
//   3. gx = xb @ W_ih0^T + b_ih0            (WMMA GEMM, M=262144, K=64,  N=768)
//   4. persistent scan layer 0 -> h1 (bf16): W_hh register-resident,
//      gx_t double-buffered in LDS via GLOBAL_LOAD_ASYNC_TO_LDS
//   5. gx = h1 @ W_ih1^T + b_ih1            (WMMA GEMM, K=256, N=768)
//   6. persistent scan layer 1 -> h2
//   7. out = h2 @ W_lin^T + b_lin + y       (WMMA GEMM, K=256, N=64)
//
// Roofline: ~2 GB HBM traffic => ~90us floor; the 2x4096-step recurrence
// dominates, so the scan keeps h in LDS, W_hh in VGPRs (24 waves x 128 VGPRs
// = 384 KB = whole bf16 W_hh), and prefetches gx_t one step ahead with the
// async-to-LDS engine so no global latency sits on the critical path.
// ---------------------------------------------------------------------------

typedef __attribute__((ext_vector_type(16))) __bf16 v16bf;
typedef __attribute__((ext_vector_type(8)))  __bf16 v8bf;
typedef __attribute__((ext_vector_type(8)))  float  v8f;
typedef __attribute__((ext_vector_type(4)))  int    v4i;

#define GRU_B   64
#define GRU_T   4096
#define GRU_I   64
#define GRU_H   256
#define GRU_G   768
#define GRU_BT  (GRU_B * GRU_T)   // 262144

#if defined(__AMDGCN__) && __has_builtin(__builtin_amdgcn_global_load_async_to_lds_b128)
#define GRU_ASYNC 1
#else
#define GRU_ASYNC 0
#endif

__device__ __forceinline__ v16bf make_frag(v8bf lo, v8bf hi) {
  return __builtin_shufflevector(lo, hi, 0, 1, 2, 3, 4, 5, 6, 7,
                                         8, 9, 10, 11, 12, 13, 14, 15);
}

__device__ __forceinline__ void async_wait0() {
#if defined(__AMDGCN__)
#if __has_builtin(__builtin_amdgcn_s_wait_asynccnt)
  __builtin_amdgcn_s_wait_asynccnt(0);
#else
  asm volatile("s_wait_asynccnt 0x0" ::: "memory");
#endif
#endif
}

#if GRU_ASYNC
// 16-byte global -> LDS async copy (tracked with ASYNCcnt, no VGPR data).
// Builtin signature (probe-confirmed): takes flat v4i* pointers + 2 imm ints.
__device__ __forceinline__ void async_copy16(const void* gsrc, void* ldst) {
  __builtin_amdgcn_global_load_async_to_lds_b128((v4i*)gsrc, (v4i*)ldst, 0, 0);
}
#endif

// ---------------------------------------------------------------------------
// Weight repack: W is [N][K] row-major fp32.  B-fragment layout for
// v_wmma_*_16x16x32_bf16: lane l -> n = l%16, kb = (l/16)*8;
// elems 0..7 = B(kt*32+kb+e, n), elems 8..15 = B(kt*32+16+kb+e, n),
// B(k,n) = W[n][k].  Each lane's 16 elements packed as 32 contiguous bytes.
// ---------------------------------------------------------------------------
__global__ void pack_weight(const float* __restrict__ W, __bf16* __restrict__ pack,
                            int K, int N) {
  int idx = blockIdx.x * blockDim.x + threadIdx.x;    // (kt*NT + nt)*32 + lane
  int total = (K >> 5) * (N >> 4) * 32;
  if (idx >= total) return;
  int lane = idx & 31;
  int tile = idx >> 5;
  int NT = N >> 4;
  int kt = tile / NT, nt = tile % NT;
  int n = lane & 15;
  int kb = (lane >> 4) << 3;
  const float* src = W + (size_t)(nt * 16 + n) * K + kt * 32;
  __bf16* dst = pack + (size_t)idx * 16;
#pragma unroll
  for (int e = 0; e < 8; ++e) {
    dst[e]     = (__bf16)src[kb + e];
    dst[8 + e] = (__bf16)src[16 + kb + e];
  }
}

__global__ void add_cvt(const float* __restrict__ y, const float* __restrict__ z,
                        __bf16* __restrict__ xb, int n) {
  int i = blockIdx.x * blockDim.x + threadIdx.x;
  if (i < n) xb[i] = (__bf16)(y[i] + z[i]);
}

// ---------------------------------------------------------------------------
// Generic WMMA GEMM: out[m,n] = sum_k A[m,k]*B(k,n) + bias[n] (+ res[m,n]).
// Block = 8 waves; each wave computes a 16(M) x 64(N) tile with 4 f32 accs.
// ---------------------------------------------------------------------------
__global__ __launch_bounds__(256) void gemm_bf16_wmma(
    const __bf16* __restrict__ A, const __bf16* __restrict__ Bp,
    const float* __restrict__ bias, const float* __restrict__ res,
    float* __restrict__ out, int M, int N, int K) {
  const int wave = threadIdx.x >> 5;
  const int lane = threadIdx.x & 31;
  const int m0 = (blockIdx.y * 8 + wave) * 16;
  const int n0 = blockIdx.x * 64;
  if (m0 >= M) return;                       // whole wave exits together
  const int NT = N >> 4;

  v8f acc[4] = {};
  const int mrow = lane & 15;
  const int kb = (lane >> 4) << 3;
  const __bf16* arow = A + (size_t)(m0 + mrow) * K + kb;

  for (int kt = 0; kt < (K >> 5); ++kt) {
    v8bf alo = *(const v8bf*)(arow);
    v8bf ahi = *(const v8bf*)(arow + 16);
    arow += 32;
    v16bf af = make_frag(alo, ahi);
#pragma unroll
    for (int j = 0; j < 4; ++j) {
      const __bf16* bp = Bp + ((size_t)(kt * NT + (n0 >> 4) + j) << 9) + lane * 16;
      v8bf blo = *(const v8bf*)bp;
      v8bf bhi = *(const v8bf*)(bp + 8);
      v16bf bf = make_frag(blo, bhi);
      acc[j] = __builtin_amdgcn_wmma_f32_16x16x32_bf16(
          false, af, false, bf, (short)0, acc[j], false, false);
    }
  }

  const int nn = lane & 15;
  const int mh = (lane >> 4) << 3;
#pragma unroll
  for (int j = 0; j < 4; ++j) {
    const int col = n0 + j * 16 + nn;
    const float bsum = bias ? bias[col] : 0.0f;
#pragma unroll
    for (int i = 0; i < 8; ++i) {
      const size_t off = (size_t)(m0 + mh + i) * N + col;
      float v = acc[j][i] + bsum;
      if (res) v += res[off];
      out[off] = v;
    }
  }
}

// ---------------------------------------------------------------------------
// Persistent GRU scan: 4 workgroups x 16 batch rows, 768 threads (24 waves).
// W_hh fragments live in VGPRs for the entire scan (wave w owns gh columns
// [32w, 32w+32): 8(kt) x 2(nt) = 16 v16bf = 128 VGPRs). Per step:
//   - issue async global->LDS copy of gx_{t+1} (double-buffered, 48 KB)
//   - gh = h @ W_hh^T : LDS A-fragments + 16 register-B WMMAs per wave
//   - fused gates from LDS-staged gx_t; h state fp32 in LDS
// Only s_barrier + s_wait_asynccnt per step; no global latency on the chain.
// ---------------------------------------------------------------------------
__global__ __launch_bounds__(768) void gru_scan(
    const float* __restrict__ gx,        // [B*T, 768], row = b*T + t
    const __bf16* __restrict__ Wp,       // packed W_hh fragments (K=256,N=768)
    const float* __restrict__ bhh,       // [768]
    __bf16* __restrict__ hseq,           // [B*T, 256] bf16, row = b*T + t
    int T) {
  __shared__ __align__(16) __bf16 h_bf[16][264];   // bf16 h (WMMA A operand)
  __shared__ float  h_f[16][256];                  // fp32 h state
  __shared__ __bf16 gh[16][768];                   // recurrent pre-activations
#if GRU_ASYNC
  extern __shared__ __align__(16) float gxs[];     // [2][16][768] staging
#endif

  const int tid = threadIdx.x;
  const int wave = tid >> 5;
  const int lane = tid & 31;
  const int bb0 = blockIdx.x * 16;

  // --- W_hh -> registers, resident across all T steps ---
  v16bf bfrag[8][2];
#pragma unroll
  for (int kt = 0; kt < 8; ++kt)
#pragma unroll
    for (int j = 0; j < 2; ++j) {
      const __bf16* bp = Wp + ((size_t)(kt * 48 + wave * 2 + j) << 9) + lane * 16;
      bfrag[kt][j] = make_frag(*(const v8bf*)bp, *(const v8bf*)(bp + 8));
    }

  for (int i = tid; i < 16 * 256; i += 768) {
    int m = i >> 8, n = i & 255;
    h_f[m][n] = 0.0f;
    h_bf[m][n] = (__bf16)0.0f;
  }
#if GRU_ASYNC
  for (int c = tid; c < 16 * 192; c += 768) {      // stage gx tile for t=0
    int m = c / 192, q = (c % 192) * 4;
    async_copy16(gx + ((size_t)(bb0 + m) * T) * GRU_G + q, &gxs[m * GRU_G + q]);
  }
  async_wait0();
#endif
  __syncthreads();

  const int n0 = wave * 32;
  for (int t = 0; t < T; ++t) {
#if GRU_ASYNC
    if (t + 1 < T) {                               // prefetch gx_{t+1} -> LDS
      float* dstbuf = gxs + ((t + 1) & 1) * (16 * GRU_G);
      for (int c = tid; c < 16 * 192; c += 768) {
        int m = c / 192, q = (c % 192) * 4;
        async_copy16(gx + ((size_t)(bb0 + m) * T + (t + 1)) * GRU_G + q,
                     dstbuf + m * GRU_G + q);
      }
    }
#endif
    // ---- phase 1: gh = h @ W_hh^T (register-resident B) ----
    v8f acc0 = {}, acc1 = {};
    const int mrow = lane & 15;
    const int kb = (lane >> 4) << 3;
#pragma unroll
    for (int kt = 0; kt < 8; ++kt) {
      const __bf16* ap = &h_bf[mrow][kt * 32 + kb];
      v16bf af = make_frag(*(const v8bf*)ap, *(const v8bf*)(ap + 16));
      acc0 = __builtin_amdgcn_wmma_f32_16x16x32_bf16(
          false, af, false, bfrag[kt][0], (short)0, acc0, false, false);
      acc1 = __builtin_amdgcn_wmma_f32_16x16x32_bf16(
          false, af, false, bfrag[kt][1], (short)0, acc1, false, false);
    }
    {
      const int nn = lane & 15;
      const int mh = (lane >> 4) << 3;
#pragma unroll
      for (int i = 0; i < 8; ++i) {
        gh[mh + i][n0 + nn]      = (__bf16)acc0[i];
        gh[mh + i][n0 + 16 + nn] = (__bf16)acc1[i];
      }
    }
    __syncthreads();

    // ---- phase 2: gates + state update (r, z, n order) ----
    for (int idx = tid; idx < 16 * 256; idx += 768) {
      const int m = idx >> 8;
      const int n = idx & 255;
      const size_t row = (size_t)(bb0 + m) * T + t;
#if GRU_ASYNC
      const float* g = gxs + (t & 1) * (16 * GRU_G) + m * GRU_G;
#else
      const float* g = gx + row * GRU_G;
#endif
      float gr = g[n]       + (float)gh[m][n]       + bhh[n];
      float gz = g[256 + n] + (float)gh[m][256 + n] + bhh[256 + n];
      float gn = g[512 + n];
      float hn = (float)gh[m][512 + n] + bhh[512 + n];
      float r = 1.0f / (1.0f + __expf(-gr));
      float zz = 1.0f / (1.0f + __expf(-gz));
      float nn = tanhf(gn + r * hn);
      float h = (1.0f - zz) * nn + zz * h_f[m][n];
      h_f[m][n] = h;
      h_bf[m][n] = (__bf16)h;
      hseq[row * GRU_H + n] = (__bf16)h;
#if !GRU_ASYNC
      if (((n & 63) == 0) && (t + 1 < T)) {
        __builtin_prefetch(gx + row * GRU_G + GRU_G + n, 0, 1);
        __builtin_prefetch(gx + row * GRU_G + GRU_G + 256 + n, 0, 1);
        __builtin_prefetch(gx + row * GRU_G + GRU_G + 512 + n, 0, 1);
      }
#endif
    }
#if GRU_ASYNC
    async_wait0();           // t+1 staging complete before it is consumed
#endif
    __syncthreads();
  }
}

// ---------------------------------------------------------------------------
extern "C" void kernel_launch(void* const* d_in, const int* in_sizes, int n_in,
                              void* d_out, int out_size, void* d_ws, size_t ws_size,
                              hipStream_t stream) {
  (void)in_sizes; (void)n_in; (void)out_size; (void)ws_size;

  const float* z     = (const float*)d_in[0];
  const float* y     = (const float*)d_in[1];
  const float* W_ih0 = (const float*)d_in[2];
  const float* W_hh0 = (const float*)d_in[3];
  const float* b_ih0 = (const float*)d_in[4];
  const float* b_hh0 = (const float*)d_in[5];
  const float* W_ih1 = (const float*)d_in[6];
  const float* W_hh1 = (const float*)d_in[7];
  const float* b_ih1 = (const float*)d_in[8];
  const float* b_hh1 = (const float*)d_in[9];
  const float* W_lin = (const float*)d_in[10];
  const float* b_lin = (const float*)d_in[11];
  float* out = (float*)d_out;

  // Workspace layout (256B aligned offsets); total ~1.11 GB.
  char* ws = (char*)d_ws;
  float*  gx  = (float*)ws;                                    // 805306368 B
  __bf16* xb  = (__bf16*)(ws + 805306368u);                    // 33554432 B
  __bf16* h1  = (__bf16*)(ws + 838860800u);                    // 134217728 B
  __bf16* h2  = (__bf16*)(ws + 973078528u);                    // 134217728 B
  __bf16* Wp_ih0 = (__bf16*)(ws + 1107296256u);                // 98304 B
  __bf16* Wp_hh0 = (__bf16*)(ws + 1107296256u + 98304u);       // 393216 B
  __bf16* Wp_ih1 = (__bf16*)(ws + 1107296256u + 491520u);      // 393216 B
  __bf16* Wp_hh1 = (__bf16*)(ws + 1107296256u + 884736u);      // 393216 B
  __bf16* Wp_lin = (__bf16*)(ws + 1107296256u + 1277952u);     // 32768 B

  pack_weight<<<dim3((2 * 48 * 32 + 255) / 256), 256, 0, stream>>>(W_ih0, Wp_ih0, GRU_I, GRU_G);
  pack_weight<<<dim3((8 * 48 * 32 + 255) / 256), 256, 0, stream>>>(W_hh0, Wp_hh0, GRU_H, GRU_G);
  pack_weight<<<dim3((8 * 48 * 32 + 255) / 256), 256, 0, stream>>>(W_ih1, Wp_ih1, GRU_H, GRU_G);
  pack_weight<<<dim3((8 * 48 * 32 + 255) / 256), 256, 0, stream>>>(W_hh1, Wp_hh1, GRU_H, GRU_G);
  pack_weight<<<dim3((8 * 4 * 32 + 255) / 256), 256, 0, stream>>>(W_lin, Wp_lin, GRU_H, GRU_I);

  add_cvt<<<dim3((GRU_BT * GRU_I + 255) / 256), 256, 0, stream>>>(y, z, xb, GRU_BT * GRU_I);

  gemm_bf16_wmma<<<dim3(GRU_G / 64, GRU_BT / 128), 256, 0, stream>>>(
      xb, Wp_ih0, b_ih0, nullptr, gx, GRU_BT, GRU_G, GRU_I);

  gru_scan<<<dim3(GRU_B / 16), 768, 98304, stream>>>(gx, Wp_hh0, b_hh0, h1, GRU_T);

  gemm_bf16_wmma<<<dim3(GRU_G / 64, GRU_BT / 128), 256, 0, stream>>>(
      h1, Wp_ih1, b_ih1, nullptr, gx, GRU_BT, GRU_G, GRU_H);

  gru_scan<<<dim3(GRU_B / 16), 768, 98304, stream>>>(gx, Wp_hh1, b_hh1, h2, GRU_T);

  gemm_bf16_wmma<<<dim3(GRU_I / 64, GRU_BT / 128), 256, 0, stream>>>(
      h2, Wp_lin, b_lin, y, out, GRU_BT, GRU_I, GRU_H);
}
